// ResASPP2_test_57715770524153
// MI455X (gfx1250) — matrix operand order repaired and checked by reading, not compiled, not run on hardware
//
#include <hip/hip_runtime.h>

// ---------------------------------------------------------------------------
// ResASPP2 on MI455X (gfx1250): bf16 WMMA implicit-GEMM convolutions.
// Activations kept in NHWC bf16 (L2-resident: ~134MB < 192MB L2), accumulate
// f32 in v_wmma_f32_16x16x32_bf16. Channel shuffles folded into the merged-
// buffer layout. BN+bias+ReLU folded into per-channel scale/shift.
// Interior tiles: fully unrolled K loop with compile-time dilation -> all B
// addresses fold into global_load immediate offsets, no exec-mask dance.
// Each wave runs two N-tiles (2 acc chains) per A fragment for WMMA ILP.
// B fragment in contiguous K order (A packed to match): lane<16 holds K=0..15, lane>=16 holds K=16..31
// (element t <-> K = t + 16*(lane>=16)) -> one contiguous 32B run per lane.
// ---------------------------------------------------------------------------

typedef __attribute__((ext_vector_type(16))) __bf16       v16bf;
typedef __attribute__((ext_vector_type(8)))  float        v8f;
typedef __attribute__((ext_vector_type(4)))  unsigned int v4u;

#define DEV static __device__ __forceinline__

DEV unsigned short f2bf(float f) {            // f32 -> bf16 round-to-nearest-even
  union { float f; unsigned u; } v; v.f = f;
  unsigned r = v.u + 0x7fffu + ((v.u >> 16) & 1u);
  return (unsigned short)(r >> 16);
}

template <int MODE> struct Cfg {
  static constexpr int CIN = (MODE == 2) ? 32 : ((MODE == 3) ? 128 : 64);
  static constexpr int KPT = CIN / 32;
  static constexpr int NK  = 9 * KPT;
  static constexpr int NC  = (MODE == 3) ? 3 : 64;
};

// ---------------------------------------------------------------------------
// Prep: per-channel BN scale/shift + weight repack into WMMA A-fragment order
// pack flat index = ((branch*NK + s)*32 + lane)*16 + t
// A layout (contiguous K, k = t + 16*(lane>=16)): lane<16 M=lane, t=0..15 -> K=0..15;
// lane>=16 M=lane-16, t=0..15 -> K=16..31.
// K ordering: k = tap*CIN + cin  (tap-major -> B runs are contiguous NHWC bytes)
// ---------------------------------------------------------------------------
struct Params {
  const float *s1w, *s1b, *g1, *b1, *m1, *v1;
  const float *s2w, *s2b, *g2, *b2, *m2, *v2;
  const float *dw,  *db,  *gd, *bd, *md, *vd;
  const float *ow,  *ob,  *go, *bo, *mo, *vo;
  unsigned short *pkS1, *pkS2, *pkS1d, *pkOut;
  float *ssS1, *ssS2, *ssS1d, *ssOut;
};

DEV void packW(int tid, unsigned short* dst, const float* w,
               int NB, int CIN, int pm /*0=dense,1=s2 block-diag,2=out pad*/) {
  const int KPT = CIN / 32, NK = 9 * KPT, total = NB * NK * 512;
  for (int idx = tid; idx < total; idx += 256) {
    const int br  = idx / (NK * 512);
    const int rem = idx % (NK * 512);
    const int s   = rem >> 9;
    const int li  = rem & 511;
    const int lane = li >> 4, t = li & 15;
    const int m  = lane & 15;
    const int r0 = (lane & 16) ? 16 : 0;
    const int kl  = r0 + t;
    const int tap = s / KPT, h = s % KPT, cin = h * 32 + kl;
    float v = 0.f;
    if (pm == 0)       v = w[((br * 16 + m) * CIN + cin) * 9 + tap];
    else if (pm == 1) { if ((m >> 2) == (cin >> 4))
                         v = w[((br * 16 + m) * 16 + (cin & 15)) * 9 + tap]; }
    else              { if (m < 3) v = w[(m * CIN + cin) * 9 + tap]; }
    dst[idx] = f2bf(v);
  }
}

__global__ __launch_bounds__(256) void prep_kernel(Params P) {
  const int tid = threadIdx.x;
  for (int q = tid; q < 64; q += 256) {
    float sc;
    sc = P.g1[q] * rsqrtf(P.v1[q] + 1e-5f);
    P.ssS1[q]  = sc; P.ssS1[64 + q]  = (P.s1b[q] - P.m1[q]) * sc + P.b1[q];
    sc = P.g2[q] * rsqrtf(P.v2[q] + 1e-5f);
    P.ssS2[q]  = sc; P.ssS2[64 + q]  = (P.s2b[q] - P.m2[q]) * sc + P.b2[q];
    sc = P.gd[q] * rsqrtf(P.vd[q] + 1e-5f);
    P.ssS1d[q] = sc; P.ssS1d[64 + q] = (P.db[q]  - P.md[q]) * sc + P.bd[q];
  }
  if (tid < 3) {
    float sc = P.go[tid] * rsqrtf(P.vo[tid] + 1e-5f);
    P.ssOut[tid] = sc; P.ssOut[3 + tid] = (P.ob[tid] - P.mo[tid]) * sc + P.bo[tid];
  }
  packW(tid, P.pkS1,  P.s1w, 4,  64, 0);
  packW(tid, P.pkS2,  P.s2w, 4,  64, 1);
  packW(tid, P.pkS1d, P.dw,  4,  32, 0);
  packW(tid, P.pkOut, P.ow,  1, 128, 2);
}

// ---------------------------------------------------------------------------
// Small direct convs (memory-bound): residual = BNReLU(conv3x3 3->3 (x)) [f32]
// hxin = BNReLU(conv3x3 3->64 (residual))  [bf16 NHWC]
// ---------------------------------------------------------------------------
__global__ __launch_bounds__(256)
void res_kernel(const float* __restrict__ x, const float* __restrict__ rw,
                const float* __restrict__ rb, const float* __restrict__ g,
                const float* __restrict__ be, const float* __restrict__ mn,
                const float* __restrict__ vr, float* __restrict__ residual) {
  const int gid = blockIdx.x * 256 + threadIdx.x;
  const int b = gid >> 16, y = (gid >> 8) & 255, xx = gid & 255;
  float patch[3][9];
  #pragma unroll
  for (int c = 0; c < 3; ++c)
    #pragma unroll
    for (int t = 0; t < 9; ++t) {
      const int yy = y + t / 3 - 1, xw = xx + t % 3 - 1;
      patch[c][t] = ((unsigned)yy < 256u && (unsigned)xw < 256u)
                    ? x[((b * 3 + c) << 16) | (yy << 8) | xw] : 0.f;
    }
  #pragma unroll
  for (int m = 0; m < 3; ++m) {
    float acc = 0.f;
    #pragma unroll
    for (int c = 0; c < 3; ++c)
      #pragma unroll
      for (int t = 0; t < 9; ++t)
        acc = fmaf(rw[(m * 3 + c) * 9 + t], patch[c][t], acc);
    const float sc = g[m] * rsqrtf(vr[m] + 1e-5f);
    float v = (acc + rb[m] - mn[m]) * sc + be[m];
    residual[((b * 3 + m) << 16) | (y << 8) | xx] = v > 0.f ? v : 0.f;
  }
}

__global__ __launch_bounds__(256)
void in_kernel(const float* __restrict__ residual, const float* __restrict__ w,
               const float* __restrict__ bia, const float* __restrict__ g,
               const float* __restrict__ be, const float* __restrict__ mn,
               const float* __restrict__ vr, unsigned short* __restrict__ hxin) {
  __shared__ float sw[1728];
  __shared__ float ssc[64], ssh[64];
  for (int i = threadIdx.x; i < 1728; i += 256) sw[i] = w[i];
  for (int q = threadIdx.x; q < 64; q += 256) {
    const float sc = g[q] * rsqrtf(vr[q] + 1e-5f);
    ssc[q] = sc; ssh[q] = (bia[q] - mn[q]) * sc + be[q];
  }
  __syncthreads();
  const int gid = blockIdx.x * 256 + threadIdx.x;
  const int b = gid >> 16, y = (gid >> 8) & 255, xx = gid & 255;
  float patch[3][9];
  #pragma unroll
  for (int c = 0; c < 3; ++c)
    #pragma unroll
    for (int t = 0; t < 9; ++t) {
      const int yy = y + t / 3 - 1, xw = xx + t % 3 - 1;
      patch[c][t] = ((unsigned)yy < 256u && (unsigned)xw < 256u)
                    ? residual[((b * 3 + c) << 16) | (yy << 8) | xw] : 0.f;
    }
  for (int m0 = 0; m0 < 64; m0 += 8) {
    union { unsigned short h[8]; v4u q; } o;
    #pragma unroll
    for (int r = 0; r < 8; ++r) {
      const int m = m0 + r;
      float acc = 0.f;
      #pragma unroll
      for (int c = 0; c < 3; ++c)
        #pragma unroll
        for (int t = 0; t < 9; ++t)
          acc = fmaf(sw[(m * 3 + c) * 9 + t], patch[c][t], acc);
      float v = acc * ssc[m] + ssh[m];
      o.h[r] = f2bf(v > 0.f ? v : 0.f);
    }
    *(v4u*)(hxin + (gid << 6) + m0) = o.q;    // NHWC, 16B aligned
  }
}

// ---------------------------------------------------------------------------
// Interior fast path: fully unrolled K loop, compile-time DIL -> all offsets
// fold into global_load_b128 immediate offsets. Two N-tiles per wave.
// B base pointers already include +16*l16 (lane-half K offset); each k-step
// reads one contiguous 32B run per lane as two adjacent b128 loads.
// ---------------------------------------------------------------------------
template <int MODE, int DIL>
DEV void fastK(const unsigned short* __restrict__ wp,
               const unsigned short* __restrict__ bp0,
               const unsigned short* __restrict__ bp1,
               const unsigned short* __restrict__ bs0,
               const unsigned short* __restrict__ bs1,
               v8f& acc0, v8f& acc1) {
  constexpr int KPT = Cfg<MODE>::KPT, NK = Cfg<MODE>::NK;
  #pragma unroll
  for (int s = 0; s < NK; ++s) {
    const int tap = s / KPT, h = s % KPT;
    const int od = ((tap / 3 - 1) * 256 + (tap % 3 - 1)) * DIL;  // pixel delta
    int oA;
    const unsigned short *p0, *p1;
    if constexpr (MODE == 0)      { oA = od * 64  + h * 32; p0 = bp0; p1 = bp1; }
    else if constexpr (MODE == 1) { oA = od * 128 + 64 * h; p0 = bp0; p1 = bp1; }
    else if constexpr (MODE == 2) { oA = od * 128;          p0 = bp0; p1 = bp1; }
    else                          { oA = od * 64 + (h & 1) * 32;
                                    if (h < 2) { p0 = bp0; p1 = bp1; }
                                    else       { p0 = bs0; p1 = bs1; } }
    v16bf a = *(const v16bf*)(wp + s * 512);
    union { v4u q[2]; v16bf v; } b0, b1;
    b0.q[0] = *(const v4u*)(p0 + oA); b0.q[1] = *(const v4u*)(p0 + oA + 8);
    b1.q[0] = *(const v4u*)(p1 + oA); b1.q[1] = *(const v4u*)(p1 + oA + 8);
    acc0 = __builtin_amdgcn_wmma_f32_16x16x32_bf16(false, a, false, b0.v,
                                                   (short)0, acc0, false, false);
    acc1 = __builtin_amdgcn_wmma_f32_16x16x32_bf16(false, a, false, b1.v,
                                                   (short)0, acc1, false, false);
  }
}

// Boundary slow path: runtime dil, per-lane zero-padded loads (one N-tile).
template <int MODE>
DEV v8f slowK(const unsigned short* __restrict__ wp,
              const unsigned short* __restrict__ hxin,
              const unsigned short* __restrict__ merged,
              const unsigned short* __restrict__ s1d,
              int b, int y, int x, int dil, int branch, int rB) {
  constexpr int KPT = Cfg<MODE>::KPT, NK = Cfg<MODE>::NK;
  v8f acc = {0.f, 0.f, 0.f, 0.f, 0.f, 0.f, 0.f, 0.f};
  for (int s = 0; s < NK; ++s) {
    v16bf a = *(const v16bf*)(wp + s * 512);
    if (s + 1 < NK) __builtin_prefetch(wp + (s + 1) * 512, 0, 0);
    const int tap = s / KPT, h = s % KPT;
    const int yy = y + (tap / 3 - 1) * dil;
    const int xx = x + (tap % 3 - 1) * dil;
    v4u qa = {0u, 0u, 0u, 0u}, qb = {0u, 0u, 0u, 0u};
    if ((unsigned)yy < 256u && (unsigned)xx < 256u) {
      const int pix = (b << 16) | (yy << 8) | xx;
      const unsigned short* pA;
      if constexpr (MODE == 0)      pA = hxin + pix * 64 + h * 32 + rB;
      else if constexpr (MODE == 1) pA = merged + pix * 128 + 64 * h + 2 * rB;
      else if constexpr (MODE == 2) pA = merged + pix * 128 + 32 * branch + rB;
      else pA = ((h < 2) ? (hxin + pix * 64) : (s1d + pix * 64))
                + (h & 1) * 32 + rB;
      qa = *(const v4u*)pA;
      qb = *(const v4u*)(pA + 8);
    }
    union { v4u q[2]; v16bf v; } bf; bf.q[0] = qa; bf.q[1] = qb;
    acc = __builtin_amdgcn_wmma_f32_16x16x32_bf16(false, a, false, bf.v,
                                                  (short)0, acc, false, false);
  }
  return acc;
}

// Epilogue: BN/bias/ReLU folded scale+shift, folded shuffle on store.
template <int MODE>
DEV void epiStore(v8f acc, int pix, int branch, int l16,
                  const float* __restrict__ ss,
                  unsigned short* __restrict__ merged,
                  unsigned short* __restrict__ s1d,
                  const float* __restrict__ residual,
                  float* __restrict__ out, int bImg) {
  constexpr int NC = Cfg<MODE>::NC;
  const float* scale = ss;
  const float* shift = ss + NC;
  if constexpr (MODE == 0 || MODE == 2) {
    union { unsigned short h[8]; v4u q; } o;
    #pragma unroll
    for (int r = 0; r < 8; ++r) {
      const int m = r + 8 * l16, q = 16 * branch + m;
      float v = acc[r] * scale[q] + shift[q];
      o.h[r] = f2bf(v > 0.f ? v : 0.f);
    }
    if constexpr (MODE == 0)
      *(v4u*)(merged + pix * 128 + 32 * branch + 8 * l16) = o.q;
    else
      *(v4u*)(s1d + pix * 64 + 16 * branch + 8 * l16) = o.q;
  } else if constexpr (MODE == 1) {
    // s2[16*branch+m] -> merged[32*(m>>2) + 16 + 4*branch + (m&3)]
    #pragma unroll
    for (int r = 0; r < 8; ++r) {
      const int m = r + 8 * l16, q = 16 * branch + m;
      float v = acc[r] * scale[q] + shift[q];
      merged[pix * 128 + 32 * (m >> 2) + 16 + 4 * branch + (m & 3)]
          = f2bf(v > 0.f ? v : 0.f);
    }
  } else {
    #pragma unroll
    for (int r = 0; r < 8; ++r) {
      const int m = r + 8 * l16;
      if (m < 3) {
        float v = acc[r] * scale[m] + shift[m];
        v = v > 0.f ? v : 0.f;
        const int oi = ((bImg * 3 + m) << 16) | (pix & 0xffff);
        out[oi] = v + residual[oi];
      }
    }
  }
}

// ---------------------------------------------------------------------------
// WMMA implicit-GEMM conv. MODE: 0=s1(hxin->merged lo), 1=s2(s1->merged hi,
// block-diag grouped), 2=s1d(merged->s1d), 3=out conv(128->3, +residual, f32)
// Block: 8 waves, one branch (uniform), 256 consecutive pixels per block;
// each wave: M=16 out-channels x 32 pixels (two WMMA N-tiles).
// ---------------------------------------------------------------------------
template <int MODE>
__global__ __launch_bounds__(256)
void gemm_conv(const unsigned short* __restrict__ wpack,
               const float* __restrict__ ss,
               const unsigned short* __restrict__ hxin,
               unsigned short* __restrict__ merged,
               unsigned short* __restrict__ s1d,
               const float* __restrict__ residual,
               float* __restrict__ out) {
  constexpr int NK = Cfg<MODE>::NK;

  const int w    = threadIdx.x >> 5;
  const int lane = threadIdx.x & 31;
  const int l16  = lane >> 4;
  const int n    = lane & 15;
  const int rB   = l16 * 16;          // B lane-half K offset (contiguous K order)

  int branch, strip;
  if constexpr (MODE == 3) { branch = 0;              strip = blockIdx.x * 256; }
  else                     { branch = blockIdx.x & 3; strip = (blockIdx.x >> 2) * 256; }
  const int pix0 = strip + w * 32;
  const int b = pix0 >> 16, y = (pix0 >> 8) & 255, xb = pix0 & 255;
  const int dil = (MODE == 3) ? 1 : (2 * branch + 1);   // DIL = 1,3,5,7

  const unsigned short* wp = wpack + branch * NK * 512 + lane * 16;
  v8f acc0 = {0.f, 0.f, 0.f, 0.f, 0.f, 0.f, 0.f, 0.f};
  v8f acc1 = acc0;

  const bool interior = (y >= dil) && (y + dil < 256) &&
                        (xb >= dil) && (xb + 31 + dil < 256);
  if (interior) {
    const int pixL = (b << 16) | (y << 8) | (xb + n);
    const unsigned short *bp0, *bp1, *bs0 = nullptr, *bs1 = nullptr;
    if constexpr (MODE == 0)      { bp0 = hxin + pixL * 64 + rB;
                                    bp1 = bp0 + 16 * 64; }
    else if constexpr (MODE == 1) { bp0 = merged + pixL * 128 + 2 * rB;
                                    bp1 = bp0 + 16 * 128; }   // s1-in-merged map
    else if constexpr (MODE == 2) { bp0 = merged + pixL * 128 + 32 * branch + rB;
                                    bp1 = bp0 + 16 * 128; }
    else                          { bp0 = hxin + pixL * 64 + rB;
                                    bp1 = bp0 + 16 * 64;
                                    bs0 = s1d + pixL * 64 + rB;
                                    bs1 = bs0 + 16 * 64; }
    if constexpr (MODE == 3)      fastK<MODE, 1>(wp, bp0, bp1, bs0, bs1, acc0, acc1);
    else if (branch == 0)         fastK<MODE, 1>(wp, bp0, bp1, bs0, bs1, acc0, acc1);
    else if (branch == 1)         fastK<MODE, 3>(wp, bp0, bp1, bs0, bs1, acc0, acc1);
    else if (branch == 2)         fastK<MODE, 5>(wp, bp0, bp1, bs0, bs1, acc0, acc1);
    else                          fastK<MODE, 7>(wp, bp0, bp1, bs0, bs1, acc0, acc1);
  } else {
    acc0 = slowK<MODE>(wp, hxin, merged, s1d, b, y, xb + n,      dil, branch, rB);
    acc1 = slowK<MODE>(wp, hxin, merged, s1d, b, y, xb + 16 + n, dil, branch, rB);
  }

  epiStore<MODE>(acc0, pix0 + n,      branch, l16, ss, merged, s1d, residual, out, b);
  epiStore<MODE>(acc1, pix0 + 16 + n, branch, l16, ss, merged, s1d, residual, out, b);
}

// ---------------------------------------------------------------------------
extern "C" void kernel_launch(void* const* d_in, const int* in_sizes, int n_in,
                              void* d_out, int out_size, void* d_ws, size_t ws_size,
                              hipStream_t stream) {
  (void)in_sizes; (void)n_in; (void)out_size; (void)ws_size;
  const float* x     = (const float*)d_in[0];
  const float* res_w = (const float*)d_in[1];
  const float* res_b = (const float*)d_in[2];
  const float *rg = (const float*)d_in[3], *rb2 = (const float*)d_in[4],
              *rm = (const float*)d_in[5], *rv  = (const float*)d_in[6];
  const float* in_w  = (const float*)d_in[7];
  const float* in_b  = (const float*)d_in[8];
  const float *ig = (const float*)d_in[9],  *ib = (const float*)d_in[10],
              *im = (const float*)d_in[11], *iv = (const float*)d_in[12];

  Params P;
  P.s1w = (const float*)d_in[13]; P.s1b = (const float*)d_in[14];
  P.g1 = (const float*)d_in[15]; P.b1 = (const float*)d_in[16];
  P.m1 = (const float*)d_in[17]; P.v1 = (const float*)d_in[18];
  P.s2w = (const float*)d_in[19]; P.s2b = (const float*)d_in[20];
  P.g2 = (const float*)d_in[21]; P.b2 = (const float*)d_in[22];
  P.m2 = (const float*)d_in[23]; P.v2 = (const float*)d_in[24];
  P.dw = (const float*)d_in[25]; P.db = (const float*)d_in[26];
  P.gd = (const float*)d_in[27]; P.bd = (const float*)d_in[28];
  P.md = (const float*)d_in[29]; P.vd = (const float*)d_in[30];
  P.ow = (const float*)d_in[31]; P.ob = (const float*)d_in[32];
  P.go = (const float*)d_in[33]; P.bo = (const float*)d_in[34];
  P.mo = (const float*)d_in[35]; P.vo = (const float*)d_in[36];

  char* ws = (char*)d_ws;
  unsigned short* hxin   = (unsigned short*)(ws + 0);           // 33,554,432 B
  unsigned short* merged = (unsigned short*)(ws + 33554432);    // 67,108,864 B
  unsigned short* s1d    = (unsigned short*)(ws + 100663296);   // 33,554,432 B
  float*          resid  = (float*)(ws + 134217728);            //  3,145,728 B
  P.pkS1  = (unsigned short*)(ws + 137363456);                  //     73,728 B
  P.pkS2  = (unsigned short*)(ws + 137437184);                  //     73,728 B
  P.pkS1d = (unsigned short*)(ws + 137510912);                  //     36,864 B
  P.pkOut = (unsigned short*)(ws + 137547776);                  //     36,864 B
  P.ssS1  = (float*)(ws + 137584640);
  P.ssS2  = (float*)(ws + 137585152);
  P.ssS1d = (float*)(ws + 137585664);
  P.ssOut = (float*)(ws + 137586176);
  float* out = (float*)d_out;

  prep_kernel<<<1, 256, 0, stream>>>(P);
  res_kernel<<<1024, 256, 0, stream>>>(x, res_w, res_b, rg, rb2, rm, rv, resid);
  in_kernel <<<1024, 256, 0, stream>>>(resid, in_w, in_b, ig, ib, im, iv, hxin);
  gemm_conv<0><<<4096, 256, 0, stream>>>(P.pkS1,  P.ssS1,  hxin, merged, s1d, resid, out);
  gemm_conv<1><<<4096, 256, 0, stream>>>(P.pkS2,  P.ssS2,  hxin, merged, s1d, resid, out);
  gemm_conv<2><<<4096, 256, 0, stream>>>(P.pkS1d, P.ssS1d, hxin, merged, s1d, resid, out);
  gemm_conv<3><<<1024, 256, 0, stream>>>(P.pkOut, P.ssOut, hxin, merged, s1d, resid, out);
}
